// NeMoCoModel_35785667510884
// MI455X (gfx1250) — matrix operand order repaired and checked by reading, not compiled, not run on hardware
//
#include <hip/hip_runtime.h>
#include <hip/hip_bf16.h>
#include <math.h>

// ---------------------------------------------------------------------------
// MoE block for MI455X (gfx1250, wave32, WMMA).
//   gating:  (B,128)->(B,8) softmax weights, fp32 scalar (tiny)
//   experts: each layer folded into ONE GEMM with K'=8*512=4096:
//              y[b,n] = sum_{e,i} (g[b,e]*x[b,i]) * alpha[e,n,i] + sum_e g[b,e]*beta[e,n]
//            A operand gate-scaled during the fp32->f16 LDS fill.
//   f16 WMMA, fp32 accumulation, LDS double-buffered with compile-time buffer
//   selection (2x unrolled k-loop + peeled tail -> branch-free inner loop).
// ---------------------------------------------------------------------------

typedef _Float16 half4 __attribute__((ext_vector_type(4)));
typedef _Float16 half8 __attribute__((ext_vector_type(8)));
typedef _Float16 v16h  __attribute__((ext_vector_type(16)));
typedef float    v8f   __attribute__((ext_vector_type(8)));

#define B_TOK   2048
#define GIN     128
#define GH      128
#define NEXP    8
#define KDIM    512                 // XIN == U == 512
#define NDIM    512
#define KTOT    (NEXP * KDIM)       // 4096 virtual K

#define BMT     128                 // block tile M
#define BNT     64                  // block tile N
#define BK      32                  // k-step (== WMMA K)
#define LDSK    40                  // padded LDS row (halves): 80B stride, conflict-free

// ---------------------------------------------------------------------------
// Gating MLP: one block per token row.
// ---------------------------------------------------------------------------
__global__ __launch_bounds__(128)
void moe_gating_kernel(const float* __restrict__ gin,
                       const float* __restrict__ Wg0, const float* __restrict__ bg0,
                       const float* __restrict__ Wg1, const float* __restrict__ bg1,
                       const float* __restrict__ Wgo, const float* __restrict__ bgo,
                       float* __restrict__ g)
{
    __shared__ float xs[GIN];
    __shared__ float h1[GH];
    __shared__ float h2[GH];
    __shared__ float logit[NEXP];

    const int b = blockIdx.x;
    const int t = threadIdx.x;

    xs[t] = gin[(size_t)b * GIN + t];
    __syncthreads();

    float s = bg0[t];
    #pragma unroll 4
    for (int i = 0; i < GIN; ++i) s += xs[i] * Wg0[i * GH + t];
    h1[t] = (s > 0.f) ? s : (expf(s) - 1.f);     // ELU
    __syncthreads();

    float s2 = bg1[t];
    #pragma unroll 4
    for (int i = 0; i < GH; ++i) s2 += h1[i] * Wg1[i * GH + t];
    h2[t] = (s2 > 0.f) ? s2 : (expf(s2) - 1.f);  // ELU
    __syncthreads();

    if (t < NEXP) {
        float s3 = bgo[t];
        #pragma unroll 4
        for (int i = 0; i < GH; ++i) s3 += h2[i] * Wgo[i * NEXP + t];
        logit[t] = s3;
    }
    __syncthreads();

    if (t == 0) {
        float mx = logit[0];
        #pragma unroll
        for (int e = 1; e < NEXP; ++e) mx = fmaxf(mx, logit[e]);
        float ex[NEXP], sum = 0.f;
        #pragma unroll
        for (int e = 0; e < NEXP; ++e) { ex[e] = expf(logit[e] - mx); sum += ex[e]; }
        float inv = 1.f / sum;
        #pragma unroll
        for (int e = 0; e < NEXP; ++e) g[(size_t)b * NEXP + e] = ex[e] * inv;
    }
}

// ---------------------------------------------------------------------------
// Fused expert layer as a single K'=4096 GEMM. 256 threads = 8 wave32.
// Wave tile 32x32 (4 WMMA / k-step); block tile 128x64; grid (8,16).
// ---------------------------------------------------------------------------
__global__ __launch_bounds__(256, 2)
void moe_expert_layer_kernel(const float* __restrict__ X,      // (B, 512)
                             const float* __restrict__ alpha,  // (8, 512, 512) n-major
                             const float* __restrict__ beta,   // (8, 512)
                             const float* __restrict__ g,      // (B, 8)
                             float* __restrict__ Y,            // (B, 512)
                             int applyElu)
{
    __shared__ __align__(16) _Float16 AsBuf[2][BMT][LDSK];   // 2 x 10KB
    __shared__ __align__(16) _Float16 BsBuf[2][BNT][LDSK];   // 2 x  5KB
    __shared__ float gs[BMT][NEXP];                          // 4KB
    __shared__ float betas[NEXP][BNT];                       // 2KB

    const int tid     = threadIdx.x;
    const int lane    = tid & 31;
    const int wave    = tid >> 5;
    const int waveM   = wave >> 1;              // 0..3 -> 32-row strips
    const int waveN   = wave & 1;               // 0..1 -> 32-col strips
    const int rowBase = blockIdx.y * BMT;
    const int colBase = blockIdx.x * BNT;

    // --- stage gating weights + beta slice ------------------------------
    #pragma unroll
    for (int p = 0; p < 4; ++p) {               // gs: 1024 floats
        int idx = tid + p * 256, m = idx >> 3, e = idx & 7;
        gs[m][e] = g[(size_t)(rowBase + m) * NEXP + e];
    }
    #pragma unroll
    for (int p = 0; p < 2; ++p) {               // betas: 512 floats
        int idx = tid + p * 256, be = idx >> 6, bn = idx & 63;
        betas[be][bn] = beta[(size_t)be * NDIM + colBase + bn];
    }
    __syncthreads();                            // gs needed by first fill

    // --- loop-invariant fill geometry (per thread) ----------------------
    const int frow = tid >> 3;                  // fill row base (+32 per chunk)
    const int fc4  = (tid & 7) << 2;            // fill col within k-step
    const float* Xrow[4];
    #pragma unroll
    for (int p = 0; p < 4; ++p)
        Xrow[p] = X + (size_t)(rowBase + frow + 32 * p) * KDIM + fc4;
    const float* Wrow[2];
    #pragma unroll
    for (int p = 0; p < 2; ++p)
        Wrow[p] = alpha + (size_t)(colBase + frow + 32 * p) * KDIM + fc4;

    // --- loop-invariant fragment geometry (ISA layouts) -----------------
    const int fm0 = waveM * 32 + (lane & 15);   // A rows, m-subtile 0
    const int kbA = (lane >> 4) << 3;           // A interleave: 0 or 8
    const int kbB = (lane >> 4) << 3;           // B interleave: 0 or 8
    const int nf0 = waveN * 32 + (lane & 15);   // B cols, n-subtile 0
    const int hi8 = (lane >> 4) << 3;           // C row offset per lane half

    v8f acc[4] = {};                            // [mi*2 + ni]
    float4 xa[4], wa[2];

    // global -> regs for k-step k0
    auto loadT = [&](int k0) {
        const int    i0   = k0 & (KDIM - 1);
        const size_t eOff = (size_t)(k0 >> 9) * (size_t)NDIM * KDIM;
        #pragma unroll
        for (int p = 0; p < 4; ++p) xa[p] = *(const float4*)(Xrow[p] + i0);
        #pragma unroll
        for (int p = 0; p < 2; ++p) wa[p] = *(const float4*)(Wrow[p] + eOff + i0);
    };
    // regs -> LDS (gate-scale A, fp32->f16)
    auto storeT = [&](_Float16 (*As)[LDSK], _Float16 (*Bs)[LDSK], int k0) {
        const int e = k0 >> 9;
        #pragma unroll
        for (int p = 0; p < 4; ++p) {
            float gv = gs[frow + 32 * p][e];
            half4 h = { (_Float16)(xa[p].x * gv), (_Float16)(xa[p].y * gv),
                        (_Float16)(xa[p].z * gv), (_Float16)(xa[p].w * gv) };
            *(half4*)&As[frow + 32 * p][fc4] = h;
        }
        #pragma unroll
        for (int p = 0; p < 2; ++p) {
            half4 h = { (_Float16)wa[p].x, (_Float16)wa[p].y,
                        (_Float16)wa[p].z, (_Float16)wa[p].w };
            *(half4*)&Bs[frow + 32 * p][fc4] = h;
        }
    };
    // LDS fragments -> 4 WMMA
    auto mmaT = [&](const _Float16 (*As)[LDSK], const _Float16 (*Bs)[LDSK]) {
        union { v16h v; half8 h[2]; } a0, a1, b0, b1;
        a0.h[0] = *(const half8*)&As[fm0][kbA];
        a0.h[1] = *(const half8*)&As[fm0][kbA + 16];
        a1.h[0] = *(const half8*)&As[fm0 + 16][kbA];
        a1.h[1] = *(const half8*)&As[fm0 + 16][kbA + 16];
        b0.h[0] = *(const half8*)&Bs[nf0][kbB];
        b0.h[1] = *(const half8*)&Bs[nf0][kbB + 16];
        b1.h[0] = *(const half8*)&Bs[nf0 + 16][kbB];
        b1.h[1] = *(const half8*)&Bs[nf0 + 16][kbB + 16];
        acc[0] = __builtin_amdgcn_wmma_f32_16x16x32_f16(false, a0.v, false, b0.v,
                                                        (short)0, acc[0], false, false);
        acc[1] = __builtin_amdgcn_wmma_f32_16x16x32_f16(false, a0.v, false, b1.v,
                                                        (short)0, acc[1], false, false);
        acc[2] = __builtin_amdgcn_wmma_f32_16x16x32_f16(false, a1.v, false, b0.v,
                                                        (short)0, acc[2], false, false);
        acc[3] = __builtin_amdgcn_wmma_f32_16x16x32_f16(false, a1.v, false, b1.v,
                                                        (short)0, acc[3], false, false);
    };

    // --- prologue: fill buffer 0 with step 0 ----------------------------
    loadT(0);
    storeT(AsBuf[0], BsBuf[0], 0);

    // --- main sweep: 2 steps per iteration, compile-time buffers --------
    for (int k0 = 0; k0 < KTOT - 2 * BK; k0 += 2 * BK) {
        __syncthreads();                        // buf0 ready
        loadT(k0 + BK);
        __builtin_prefetch(Wrow[0] + (size_t)((k0 + 2 * BK) >> 9) * NDIM * KDIM
                                   + ((k0 + 2 * BK) & (KDIM - 1)), 0, 1);
        mmaT(AsBuf[0], BsBuf[0]);               // step k0
        storeT(AsBuf[1], BsBuf[1], k0 + BK);

        __syncthreads();                        // buf1 ready
        loadT(k0 + 2 * BK);
        mmaT(AsBuf[1], BsBuf[1]);               // step k0+BK
        storeT(AsBuf[0], BsBuf[0], k0 + 2 * BK);
    }

    // --- peeled tail: steps KTOT-2BK (buf0) and KTOT-BK (buf1) ----------
    __syncthreads();
    loadT(KTOT - BK);
    mmaT(AsBuf[0], BsBuf[0]);
    storeT(AsBuf[1], BsBuf[1], KTOT - BK);
    __syncthreads();
    mmaT(AsBuf[1], BsBuf[1]);

    // --- epilogue: gated bias, optional ELU, fp32 store ------------------
    const int nl0 = waveN * 32 + (lane & 15);
    const int nl1 = nl0 + 16;
    #pragma unroll
    for (int mi = 0; mi < 2; ++mi) {
        #pragma unroll
        for (int r = 0; r < 8; ++r) {
            int mloc = waveM * 32 + mi * 16 + hi8 + r;
            float b0v = 0.f, b1v = 0.f;
            #pragma unroll
            for (int e = 0; e < NEXP; ++e) {
                float gv = gs[mloc][e];
                b0v += gv * betas[e][nl0];
                b1v += gv * betas[e][nl1];
            }
            float v0 = acc[mi * 2 + 0][r] + b0v;
            float v1 = acc[mi * 2 + 1][r] + b1v;
            if (applyElu) {
                v0 = (v0 > 0.f) ? v0 : (expf(v0) - 1.f);
                v1 = (v1 > 0.f) ? v1 : (expf(v1) - 1.f);
            }
            size_t rowOff = (size_t)(rowBase + mloc) * NDIM;
            Y[rowOff + colBase + nl0] = v0;
            Y[rowOff + colBase + nl1] = v1;
        }
    }
}

// ---------------------------------------------------------------------------
// Launch: gating -> layer0 -> layer1 -> layer2 (stream-ordered).
// Workspace: [0, 64KB) gating weights g; [1MB, 5MB) actA; [5MB, 9MB) actB.
// ---------------------------------------------------------------------------
extern "C" void kernel_launch(void* const* d_in, const int* in_sizes, int n_in,
                              void* d_out, int out_size, void* d_ws, size_t ws_size,
                              hipStream_t stream)
{
    (void)in_sizes; (void)n_in; (void)out_size; (void)ws_size;

    const float* gating_input = (const float*)d_in[0];
    const float* expert_input = (const float*)d_in[1];
    const float* Wg0 = (const float*)d_in[2];
    const float* bg0 = (const float*)d_in[3];
    const float* Wg1 = (const float*)d_in[4];
    const float* bg1 = (const float*)d_in[5];
    const float* Wgo = (const float*)d_in[6];
    const float* bgo = (const float*)d_in[7];
    const float* alpha0 = (const float*)d_in[8];
    const float* beta0  = (const float*)d_in[9];
    const float* alpha1 = (const float*)d_in[10];
    const float* beta1  = (const float*)d_in[11];
    const float* alpha2 = (const float*)d_in[12];
    const float* beta2  = (const float*)d_in[13];

    float* g    = (float*)d_ws;                                    // 64KB
    float* actA = (float*)((char*)d_ws + (1u << 20));              // 4MB
    float* actB = (float*)((char*)d_ws + (1u << 20) + (4u << 20)); // 4MB
    float* out  = (float*)d_out;

    // 1) gating weights
    moe_gating_kernel<<<B_TOK, 128, 0, stream>>>(gating_input, Wg0, bg0, Wg1, bg1,
                                                 Wgo, bgo, g);

    // 2) expert layers: grid = (512/64, 2048/128) = (8, 16)
    dim3 grid(NDIM / BNT, B_TOK / BMT);
    dim3 block(256);
    moe_expert_layer_kernel<<<grid, block, 0, stream>>>(expert_input, alpha0, beta0,
                                                        g, actA, 1);
    moe_expert_layer_kernel<<<grid, block, 0, stream>>>(actA, alpha1, beta1,
                                                        g, actB, 1);
    moe_expert_layer_kernel<<<grid, block, 0, stream>>>(actB, alpha2, beta2,
                                                        g, out, 0);
}
